// CrossAttention_43061342110469
// MI455X (gfx1250) — matrix operand order repaired and checked by reading, not compiled, not run on hardware
//
#include <hip/hip_runtime.h>

typedef __attribute__((ext_vector_type(2))) float v2f;
typedef __attribute__((ext_vector_type(4))) float v4f;
typedef __attribute__((ext_vector_type(8))) float v8f;
typedef __attribute__((ext_vector_type(8))) __bf16 v8bf;
typedef __attribute__((ext_vector_type(16))) __bf16 v16bf;
typedef __attribute__((ext_vector_type(4))) unsigned short v4us;
typedef __attribute__((ext_vector_type(8))) unsigned short v8us;

__device__ __forceinline__ unsigned short f32_to_bf16_rne(float f) {
  unsigned int u = __builtin_bit_cast(unsigned int, f);
  u += 0x7FFFu + ((u >> 16) & 1u);
  return (unsigned short)(u >> 16);
}
__device__ __forceinline__ float bf16_bits_to_f32(unsigned short h) {
  return __builtin_bit_cast(float, (unsigned int)h << 16);
}

// ---------------------------------------------------------------------------
// Kernel 1: Wf = Wv @ Wo (exact f32 WMMA), written TRANSPOSED as bf16 hi/lo:
//   WT_hi/WT_lo[n*D + k] with Wf[k,n] = hi + lo (hi = bf16(v), lo = bf16(v-hi))
// Wave tile 64x64 (4x4 tiles of 16x16), 2x2 waves per block => 128x128.
// f32 WMMA frag map: A/B lane holds k = 2*(lane>>4)+comp, m/n = lane&15;
// C/D lane holds m = 8*(lane>>4)+comp, n = lane&15.
// ---------------------------------------------------------------------------
__global__ __launch_bounds__(128) void wfuse_f32_to_bf16t(
    const float* __restrict__ Wv, const float* __restrict__ Wo,
    unsigned short* __restrict__ WT_hi, unsigned short* __restrict__ WT_lo,
    int D) {
  const int lane = threadIdx.x & 31;
  const int wave = threadIdx.x >> 5;
  const int m0 = blockIdx.x * 128 + (wave & 1) * 64;
  const int n0 = blockIdx.y * 128 + (wave >> 1) * 64;

  const int fr = lane & 15;
  const int fk = (lane >> 4) << 1;

  v8f acc[4][4];
#pragma unroll
  for (int i = 0; i < 4; ++i)
#pragma unroll
    for (int j = 0; j < 4; ++j) {
      v8f z = {};
      acc[i][j] = z;
    }

  const float* pA[4];
#pragma unroll
  for (int i = 0; i < 4; ++i)
    pA[i] = Wv + (size_t)(m0 + i * 16 + fr) * D + fk;
  const float* pB = Wo + (size_t)fk * D + (n0 + fr);

  for (int k0 = 0; k0 < D; k0 += 4) {
    v2f a[4], b[4];
#pragma unroll
    for (int i = 0; i < 4; ++i) a[i] = *(const v2f*)(pA[i] + k0);
#pragma unroll
    for (int j = 0; j < 4; ++j) {
      const float* p = pB + (size_t)k0 * D + j * 16;
      b[j].x = p[0];
      b[j].y = p[D];
    }
#pragma unroll
    for (int i = 0; i < 4; ++i)
#pragma unroll
      for (int j = 0; j < 4; ++j)
        acc[i][j] = __builtin_amdgcn_wmma_f32_16x16x4_f32(
            false, a[i], false, b[j], (short)0, acc[i][j], false, false);
  }

  const int cn = lane & 15;
  const int cm = (lane >> 4) << 3;
#pragma unroll
  for (int j = 0; j < 4; ++j) {
    const int col = n0 + j * 16 + cn;  // n index of Wf
#pragma unroll
    for (int i = 0; i < 4; ++i) {
      const int row = m0 + i * 16 + cm;  // k index of Wf (8 consecutive)
      v8us hi, lo;
#pragma unroll
      for (int c = 0; c < 8; ++c) {
        float f = acc[i][j][c];
        unsigned short h = f32_to_bf16_rne(f);
        hi[c] = h;
        lo[c] = f32_to_bf16_rne(f - bf16_bits_to_f32(h));
      }
      size_t base = (size_t)col * D + row;  // transposed, 16B aligned
      *(v8us*)(WT_hi + base) = hi;
      *(v8us*)(WT_lo + base) = lo;
    }
  }
}

// ---------------------------------------------------------------------------
// Kernel 2: Xs = x * 2048 (exact: 2^11), split into bf16 hi/lo planes.
// ---------------------------------------------------------------------------
__global__ __launch_bounds__(256) void split_scale_x(
    const float* __restrict__ X, unsigned short* __restrict__ Xh,
    unsigned short* __restrict__ Xl, float scale) {
  size_t i = (size_t)blockIdx.x * 256 + threadIdx.x;  // one float4 per thread
  v4f f = ((const v4f*)X)[i];
  v4us hi, lo;
#pragma unroll
  for (int c = 0; c < 4; ++c) {
    float v = f[c] * scale;
    unsigned short h = f32_to_bf16_rne(v);
    hi[c] = h;
    lo[c] = f32_to_bf16_rne(v - bf16_bits_to_f32(h));
  }
  ((v4us*)Xh)[i] = hi;
  ((v4us*)Xl)[i] = lo;
}

// ---------------------------------------------------------------------------
// Kernel 3: bfused[n] = 2048 * sum_k bv[k]*Wo[k,n] + bo[n]   (tiny GEMV)
// ---------------------------------------------------------------------------
__global__ __launch_bounds__(256) void fuse_bias(
    const float* __restrict__ bv, const float* __restrict__ Wo,
    const float* __restrict__ bo, float* __restrict__ bfused,
    int D, float scale) {
  const int n = blockIdx.x * 256 + threadIdx.x;
  float acc = 0.0f;
#pragma unroll 4
  for (int k = 0; k < D; ++k) acc = fmaf(bv[k], Wo[(size_t)k * D + n], acc);
  bfused[n] = fmaf(acc, scale, bo[n]);
}

// ---------------------------------------------------------------------------
// Kernel 4: out = Xs @ Wf + bfused, bf16 3-product split, fp32 accumulate.
//   A = Xh/Xl [M,K] row-major bf16; B = WT_hi/WT_lo [N,K] (Wf transposed).
// Wave tile 64x32 (4 m-tiles x 2 n-tiles); 8 waves (2m x 4n) => 128x128 block.
// bf16 WMMA frag map (ISA 7.12.2):
//   A lane(m=lane&15, hh=lane>>4): comps0-7 -> K=8hh+0..7, comps8-15 -> K=16+8hh+0..7
//   B lane(n=lane&15, hh):         comps0-7 -> K=8hh+0..7, comps8-15 -> K=16+8hh+0..7
//   C/D lane: m = 8hh+comp, n = lane&15
// ---------------------------------------------------------------------------
__global__ __launch_bounds__(256) void gemm_bf16_split(
    const unsigned short* __restrict__ Ah, const unsigned short* __restrict__ Al,
    const unsigned short* __restrict__ Bh, const unsigned short* __restrict__ Bl,
    const float* __restrict__ bias, float* __restrict__ Out,
    int M, int N, int K) {
  const int lane = threadIdx.x & 31;
  const int wave = threadIdx.x >> 5;  // 0..7
  const int m0 = blockIdx.x * 128 + (wave & 1) * 64;
  const int n0 = blockIdx.y * 128 + (wave >> 1) * 32;

  const int fr = lane & 15;
  const int hh = lane >> 4;

  v8f acc[4][2];
#pragma unroll
  for (int i = 0; i < 4; ++i)
#pragma unroll
    for (int j = 0; j < 2; ++j) {
      v8f z = {};
      acc[i][j] = z;
    }

  const unsigned short* pAh[4];
  const unsigned short* pAl[4];
#pragma unroll
  for (int i = 0; i < 4; ++i) {
    size_t off = (size_t)(m0 + i * 16 + fr) * K + 8 * hh;
    pAh[i] = Ah + off;
    pAl[i] = Al + off;
  }
  const unsigned short* pBh[2];
  const unsigned short* pBl[2];
#pragma unroll
  for (int j = 0; j < 2; ++j) {
    size_t off = (size_t)(n0 + j * 16 + fr) * K + 8 * hh;
    pBh[j] = Bh + off;
    pBl[j] = Bl + off;
  }

  for (int k0 = 0; k0 < K; k0 += 32) {
    v16bf ah[4], al[4], bh[2], bl[2];
#pragma unroll
    for (int i = 0; i < 4; ++i) {
      v8bf h0 = *(const v8bf*)(pAh[i] + k0);
      v8bf h1 = *(const v8bf*)(pAh[i] + k0 + 16);
      ah[i] = __builtin_shufflevector(h0, h1, 0, 1, 2, 3, 4, 5, 6, 7, 8, 9, 10,
                                      11, 12, 13, 14, 15);
      v8bf l0 = *(const v8bf*)(pAl[i] + k0);
      v8bf l1 = *(const v8bf*)(pAl[i] + k0 + 16);
      al[i] = __builtin_shufflevector(l0, l1, 0, 1, 2, 3, 4, 5, 6, 7, 8, 9, 10,
                                      11, 12, 13, 14, 15);
    }
#pragma unroll
    for (int j = 0; j < 2; ++j) {
      v8bf h0 = *(const v8bf*)(pBh[j] + k0);
      v8bf h1 = *(const v8bf*)(pBh[j] + k0 + 16);
      bh[j] = __builtin_shufflevector(h0, h1, 0, 1, 2, 3, 4, 5, 6, 7, 8, 9, 10,
                                      11, 12, 13, 14, 15);
      v8bf l0 = *(const v8bf*)(pBl[j] + k0);
      v8bf l1 = *(const v8bf*)(pBl[j] + k0 + 16);
      bl[j] = __builtin_shufflevector(l0, l1, 0, 1, 2, 3, 4, 5, 6, 7, 8, 9, 10,
                                      11, 12, 13, 14, 15);
    }
#pragma unroll
    for (int i = 0; i < 4; ++i)
#pragma unroll
      for (int j = 0; j < 2; ++j) {
        acc[i][j] = __builtin_amdgcn_wmma_f32_16x16x32_bf16(
            false, ah[i], false, bh[j], (short)0, acc[i][j], false, false);
        acc[i][j] = __builtin_amdgcn_wmma_f32_16x16x32_bf16(
            false, ah[i], false, bl[j], (short)0, acc[i][j], false, false);
        acc[i][j] = __builtin_amdgcn_wmma_f32_16x16x32_bf16(
            false, al[i], false, bh[j], (short)0, acc[i][j], false, false);
      }
  }

  const int cn = lane & 15;
  const int cm = hh << 3;
#pragma unroll
  for (int j = 0; j < 2; ++j) {
    const int col = n0 + j * 16 + cn;
    const float bval = bias[col];
#pragma unroll
    for (int i = 0; i < 4; ++i)
#pragma unroll
      for (int c = 0; c < 8; ++c) {
        const int row = m0 + i * 16 + cm + c;
        Out[(size_t)row * N + col] = acc[i][j][c] + bval;
      }
  }
}

extern "C" void kernel_launch(void* const* d_in, const int* in_sizes, int n_in,
                              void* d_out, int out_size, void* d_ws, size_t ws_size,
                              hipStream_t stream) {
  // inputs: 0:x 1:encoder_x 2:Wq 3:bq 4:Wk 5:bk 6:Wv 7:bv 8:Wo 9:bo
  // Math: softmax rows sum to 1 => einsum 'bvhd,bhqk->bvhd' multiplies v by
  // sum_{q,k} scores = L = 2048 per (b,h) (exact to ~1e-7).
  //   out = L*(x@Wv + bv)@Wo + bo = x_scaled @ (Wv@Wo) + (L*bv@Wo + bo)
  // with x_scaled = x * 2^11 (exact). Main GEMM runs as 3-product bf16 split.
  const float* x  = (const float*)d_in[0];
  const float* Wv = (const float*)d_in[6];
  const float* bv = (const float*)d_in[7];
  const float* Wo = (const float*)d_in[8];
  const float* bo = (const float*)d_in[9];
  float* out = (float*)d_out;

  const int Bb = 4, L = 2048, D = 1024;
  const int M = Bb * L;  // 8192
  const float Ls = (float)L;

  char* ws = (char*)d_ws;
  unsigned short* WT_hi = (unsigned short*)(ws);                      // 2 MB
  unsigned short* WT_lo = (unsigned short*)(ws + (2ull << 20));       // 2 MB
  float* bfused         = (float*)(ws + (4ull << 20));                // 4 KB
  unsigned short* Xh    = (unsigned short*)(ws + (8ull << 20));       // 16 MB
  unsigned short* Xl    = (unsigned short*)(ws + (24ull << 20));      // 16 MB

  // Wf = Wv @ Wo  (f32 WMMA), stored transposed bf16 hi/lo
  wfuse_f32_to_bf16t<<<dim3(D / 128, D / 128), 128, 0, stream>>>(
      Wv, Wo, WT_hi, WT_lo, D);
  // x * 2048 split into bf16 hi/lo
  split_scale_x<<<(M * D / 4) / 256, 256, 0, stream>>>(x, Xh, Xl, Ls);
  // bfused = 2048 * bv@Wo + bo
  fuse_bias<<<D / 256, 256, 0, stream>>>(bv, Wo, bo, bfused, D, Ls);
  // out = Xs @ Wf + bfused  (bf16 split WMMA)
  gemm_bf16_split<<<dim3(M / 128, D / 128), 256, 0, stream>>>(
      Xh, Xl, WT_hi, WT_lo, bfused, out, M, D, D);
}
